// ViTBlockAdaLNZero_59914793779988
// MI455X (gfx1250) — matrix operand order repaired and checked by reading, not compiled, not run on hardware
//
#include <hip/hip_runtime.h>
#include <hip/hip_bf16.h>
#include <math.h>

// ---------------------------------------------------------------------------
// DiT / AdaLN-Zero block for MI455X (gfx1250, wave32, WMMA bf16 16x16x32).
// Compute-bound (~142 GFLOP vs <100MB traffic) -> everything on the matrix
// pipe: activations & weights converted to bf16, f32 accumulation.
// GEMM: BM=128 BN=128 BK=32, double-buffered LDS (1 barrier / K-step),
// 8 WMMAs per wave per K-step, global_prefetch 2 tiles ahead.
// ---------------------------------------------------------------------------

typedef __attribute__((ext_vector_type(16))) __bf16 v16bf;
typedef __attribute__((ext_vector_type(8)))  float  v8f;

#define C_DIM 768
#define NSEQ  1024
#define BATCH 8
#define NHEAD 12
#define DHEAD 64
#define HFF   3072
#define ADA_W (6*C_DIM)
#define MROWS (BATCH*NSEQ)

#define WMMA_BF16(a,b,c) \
  __builtin_amdgcn_wmma_f32_16x16x32_bf16(false,(a),false,(b),(short)0,(c),false,false)

// ---- WMMA fragment loaders (contiguous k order k = 16*(l/16) + s, wave32) -------------------
static __device__ __forceinline__ v16bf frag_pack(const __bf16* p0, const __bf16* p1) {
  union { uint4 u[2]; v16bf v; } u;
  u.u[0] = *(const uint4*)p0;
  u.u[1] = *(const uint4*)p1;
  return u.v;
}
// A operand 16x32 (MxK): lane l -> row (l&15); halves K=0..15 (lanes 0-15)
// or K=16..31 (lanes 16-31), contiguous.
static __device__ __forceinline__ v16bf load_frag_a(const __bf16* base, int ld) {
  int lane = threadIdx.x & 31;
  const __bf16* r = base + (lane & 15) * ld;
  int h16 = (lane >> 4) * 16;
  return frag_pack(r + h16, r + h16 + 8);
}
// B operand 32x16 (KxN): lane l -> column (l&15) = memory row (N,K layout);
// halves K=0..15 (lanes 0-15) or K=16..31 (lanes 16-31), contiguous.
static __device__ __forceinline__ v16bf load_frag_b(const __bf16* base, int ld) {
  int lane = threadIdx.x & 31;
  const __bf16* r = base + (lane & 15) * ld + (lane >> 4) * 16;
  return frag_pack(r, r + 8);
}

static __device__ __forceinline__ float redmax16(float v) {
  v = fmaxf(v, __shfl_xor(v, 1, 32));
  v = fmaxf(v, __shfl_xor(v, 2, 32));
  v = fmaxf(v, __shfl_xor(v, 4, 32));
  v = fmaxf(v, __shfl_xor(v, 8, 32));
  return v;
}
static __device__ __forceinline__ float redsum16(float v) {
  v += __shfl_xor(v, 1, 32);
  v += __shfl_xor(v, 2, 32);
  v += __shfl_xor(v, 4, 32);
  v += __shfl_xor(v, 8, 32);
  return v;
}

// ---- fp32 -> bf16 weight conversion ---------------------------------------
__global__ void cvt_bf16_kernel(const float* __restrict__ s, __bf16* __restrict__ d, int n) {
  int i = (blockIdx.x * 256 + threadIdx.x) * 4;
  if (i + 3 < n) {
    float4 f = *(const float4*)(s + i);
    d[i + 0] = (__bf16)f.x; d[i + 1] = (__bf16)f.y;
    d[i + 2] = (__bf16)f.z; d[i + 3] = (__bf16)f.w;
  }
}

// ---- ada = silu(c) @ ada_w.T + ada_b (tiny: 28 MFLOP, VALU is fine) -------
__global__ void ada_kernel(const float* __restrict__ c, const float* __restrict__ w,
                           const float* __restrict__ bias, float* __restrict__ ada) {
  __shared__ float sc[C_DIM];
  int b = blockIdx.y;
  for (int i = threadIdx.x; i < C_DIM; i += 256) {
    float v = c[b * C_DIM + i];
    sc[i] = v / (1.f + __expf(-v));
  }
  __syncthreads();
  int j = blockIdx.x * 256 + threadIdx.x;
  const float* wr = w + (size_t)j * C_DIM;
  float acc = 0.f;
  for (int k = 0; k < C_DIM; ++k) acc += sc[k] * wr[k];
  ada[b * ADA_W + j] = acc + bias[j];
}

// ---- LayerNorm + AdaLN modulate -> bf16 activations -----------------------
__global__ void ln_mod_kernel(const float* __restrict__ x, const float* __restrict__ ada,
                              int sh_off, int sc_off, __bf16* __restrict__ h) {
  __shared__ float red[256];
  int r = blockIdx.x;
  int b = r / NSEQ;
  const float* xr = x + (size_t)r * C_DIM;
  int t = threadIdx.x;
  float v0 = xr[t], v1 = xr[t + 256], v2 = xr[t + 512];
  red[t] = v0 + v1 + v2;
  __syncthreads();
  for (int o = 128; o > 0; o >>= 1) { if (t < o) red[t] += red[t + o]; __syncthreads(); }
  float mu = red[0] * (1.f / 768.f);
  __syncthreads();
  float d0 = v0 - mu, d1 = v1 - mu, d2 = v2 - mu;
  red[t] = d0 * d0 + d1 * d1 + d2 * d2;
  __syncthreads();
  for (int o = 128; o > 0; o >>= 1) { if (t < o) red[t] += red[t + o]; __syncthreads(); }
  float rstd = rsqrtf(red[0] * (1.f / 768.f) + 1e-6f);
  const float* shp = ada + b * ADA_W + sh_off;
  const float* scp = ada + b * ADA_W + sc_off;
  __bf16* hr = h + (size_t)r * C_DIM;
  hr[t]       = (__bf16)(d0 * rstd * (1.f + scp[t])       + shp[t]);
  hr[t + 256] = (__bf16)(d1 * rstd * (1.f + scp[t + 256]) + shp[t + 256]);
  hr[t + 512] = (__bf16)(d2 * rstd * (1.f + scp[t + 512]) + shp[t + 512]);
}

// ---- WMMA GEMM: out = A(MxK,bf16) @ W(NxK,bf16)^T + bias, fused epilogues -
// Tile: BM=128, BN=128, BK=32; 8 waves as 4(M) x 2(N); wave tile 32x64.
// Double-buffered LDS: one barrier per K-step, global loads for tile k+1
// issued before computing tile k, prefetch for tile k+2.
// MODE 0: qkv split (q scaled by 1/8, v stored transposed (B,H,D,N))
// MODE 1: out_f32 = res + gate * (acc+bias)         (proj / fc2 residual)
// MODE 2: out_bf16 = gelu_exact(acc+bias)           (fc1)
template <int MODE>
__launch_bounds__(256)
__global__ void gemm_kernel(const __bf16* __restrict__ A, const __bf16* __restrict__ W,
                            const float* __restrict__ bias, int N, int K,
                            float* __restrict__ out_f32, __bf16* __restrict__ out_bf16,
                            const float* __restrict__ res, const float* __restrict__ gate,
                            __bf16* __restrict__ qout, __bf16* __restrict__ kout,
                            __bf16* __restrict__ vtout) {
  // 80-byte LDS row stride: 16 consecutive rows hit 16 distinct bank groups.
  __shared__ __align__(16) __bf16 sA[2 * 128 * 40];
  __shared__ __align__(16) __bf16 sB[2 * 128 * 40];
  int t = threadIdx.x;
  int lane = t & 31, w = t >> 5;
  int wm = w & 3, wn = w >> 2;            // 4x2 waves -> BM=128, BN=128
  int m0 = blockIdx.y * 128, n0 = blockIdx.x * 128;

  v8f acc[2][4] = {};

  // Per-thread staging: each thread moves one 32B chunk of A and of B.
  const __bf16* Ag = A + (size_t)(m0 + (t >> 1)) * K + (t & 1) * 16;
  const __bf16* Wg = W + (size_t)(n0 + (t >> 1)) * K + (t & 1) * 16;
  int sofs = (t >> 1) * 40 + (t & 1) * 16;

  {  // preload tile 0 into buffer 0
    uint4 a0 = *(const uint4*)(Ag);     uint4 a1 = *(const uint4*)(Ag + 8);
    uint4 b0 = *(const uint4*)(Wg);     uint4 b1 = *(const uint4*)(Wg + 8);
    *(uint4*)&sA[sofs] = a0;  *(uint4*)&sA[sofs + 8] = a1;
    *(uint4*)&sB[sofs] = b0;  *(uint4*)&sB[sofs + 8] = b1;
  }

  int nt = K >> 5;
  for (int it = 0; it < nt; ++it) {
    uint4 na0, na1, nb0, nb1;
    if (it + 1 < nt) {
      int k1 = (it + 1) << 5;
      na0 = *(const uint4*)(Ag + k1);     na1 = *(const uint4*)(Ag + k1 + 8);
      nb0 = *(const uint4*)(Wg + k1);     nb1 = *(const uint4*)(Wg + k1 + 8);
    }
    if (it + 2 < nt) {
      int k2 = (it + 2) << 5;
      __builtin_prefetch(Ag + k2, 0, 0);  // global_prefetch_b8
      __builtin_prefetch(Wg + k2, 0, 0);
    }
    __syncthreads();   // buffer (it&1) fully written by previous iteration
    const __bf16* cA = sA + (it & 1) * (128 * 40);
    const __bf16* cB = sB + (it & 1) * (128 * 40);
    v16bf af0 = load_frag_a(cA + (wm * 32) * 40, 40);
    v16bf af1 = load_frag_a(cA + (wm * 32 + 16) * 40, 40);
#pragma unroll
    for (int nf = 0; nf < 4; ++nf) {
      v16bf bfr = load_frag_b(cB + (wn * 64 + nf * 16) * 40, 40);
      acc[0][nf] = WMMA_BF16(af0, bfr, acc[0][nf]);
      acc[1][nf] = WMMA_BF16(af1, bfr, acc[1][nf]);
    }
    if (it + 1 < nt) {  // fill the other buffer; next top-of-loop barrier covers it
      __bf16* nA = sA + ((it + 1) & 1) * (128 * 40);
      __bf16* nB = sB + ((it + 1) & 1) * (128 * 40);
      *(uint4*)&nA[sofs] = na0;  *(uint4*)&nA[sofs + 8] = na1;
      *(uint4*)&nB[sofs] = nb0;  *(uint4*)&nB[sofs + 8] = nb1;
    }
  }

  int half = lane >> 4, nl = lane & 15;
  for (int mf = 0; mf < 2; ++mf)
    for (int nf = 0; nf < 4; ++nf)
      for (int vr = 0; vr < 8; ++vr) {
        int row = m0 + wm * 32 + mf * 16 + vr + 8 * half;   // C layout: m=vr+8*half
        int col = n0 + wn * 64 + nf * 16 + nl;              //           n=lane&15
        float v = acc[mf][nf][vr] + bias[col];
        if (MODE == 0) {
          int b = row / NSEQ, n = row % NSEQ;
          int s = col / C_DIM, rem = col % C_DIM;
          int hh = rem >> 6, d = rem & 63;
          size_t bh = (size_t)(b * NHEAD + hh);
          if (s == 0)      qout[(bh * NSEQ + n) * DHEAD + d] = (__bf16)(v * 0.125f);
          else if (s == 1) kout[(bh * NSEQ + n) * DHEAD + d] = (__bf16)v;
          else             vtout[bh * DHEAD * NSEQ + (size_t)d * NSEQ + n] = (__bf16)v;
        } else if (MODE == 1) {
          int b = row / NSEQ;
          out_f32[(size_t)row * N + col] =
              res[(size_t)row * N + col] + gate[b * ADA_W + col] * v;
        } else {
          out_bf16[(size_t)row * N + col] =
              (__bf16)(0.5f * v * (1.f + erff(v * 0.70710678118f)));
        }
      }
}

// ---- Flash attention: 1 wave per (b, h, 32-query block) -------------------
__launch_bounds__(32)
__global__ void attn_kernel(const __bf16* __restrict__ qb, const __bf16* __restrict__ kb,
                            const __bf16* __restrict__ vtb, __bf16* __restrict__ obuf) {
  __shared__ __align__(16) __bf16 sP[32 * 72];   // P tile 32x64, stride 72 halves
  int blk = blockIdx.x;
  int qblk = blk & 31;
  int bh = blk >> 5;
  int b = bh / NHEAD, hh = bh % NHEAD;
  int lane = threadIdx.x & 31;
  int half = lane >> 4, nl = lane & 15;

  // Q fragments held in registers for the whole row-block (q pre-scaled).
  const __bf16* qbase = qb + ((size_t)bh * NSEQ + qblk * 32) * DHEAD;
  v16bf qa[2][2];
  for (int mf = 0; mf < 2; ++mf)
    for (int ks = 0; ks < 2; ++ks)
      qa[mf][ks] = load_frag_a(qbase + mf * 16 * DHEAD + ks * 32, DHEAD);

  v8f of[2][4] = {};
  float rm[2][8], rl[2][8];
  for (int mf = 0; mf < 2; ++mf)
    for (int vr = 0; vr < 8; ++vr) { rm[mf][vr] = -1e30f; rl[mf][vr] = 0.f; }

  for (int kt = 0; kt < NSEQ / 64; ++kt) {
    // S = Q @ K^T  (K rows are (N,D) row-major == B operand layout)
    const __bf16* kbase = kb + ((size_t)bh * NSEQ + kt * 64) * DHEAD;
    v8f sf[2][4] = {};
    for (int ks = 0; ks < 2; ++ks)
      for (int nf = 0; nf < 4; ++nf) {
        v16bf bfr = load_frag_b(kbase + nf * 16 * DHEAD + ks * 32, DHEAD);
        sf[0][nf] = WMMA_BF16(qa[0][ks], bfr, sf[0][nf]);
        sf[1][nf] = WMMA_BF16(qa[1][ks], bfr, sf[1][nf]);
      }
    // Online softmax: rows live at m = vr + 8*half, cols across 16 lanes.
    for (int mf = 0; mf < 2; ++mf)
      for (int vr = 0; vr < 8; ++vr) {
        float tm = sf[mf][0][vr];
        tm = fmaxf(tm, sf[mf][1][vr]);
        tm = fmaxf(tm, sf[mf][2][vr]);
        tm = fmaxf(tm, sf[mf][3][vr]);
        tm = redmax16(tm);
        float mn = fmaxf(rm[mf][vr], tm);
        float corr = __expf(rm[mf][vr] - mn);
        rm[mf][vr] = mn;
        float ls = 0.f;
        int prow = (mf * 16 + vr + 8 * half) * 72 + nl;
        for (int nf = 0; nf < 4; ++nf) {
          float p = __expf(sf[mf][nf][vr] - mn);
          ls += p;
          sP[prow + nf * 16] = (__bf16)p;       // stage P for A-operand reload
        }
        for (int df = 0; df < 4; ++df) of[mf][df][vr] *= corr;
        rl[mf][vr] = rl[mf][vr] * corr + ls;
      }
    // O += P @ V   (V stored transposed (D,N): rows d == B-operand rows)
    const __bf16* vbase = vtb + (size_t)bh * DHEAD * NSEQ + kt * 64;
    for (int ks = 0; ks < 2; ++ks) {
      v16bf pa0 = load_frag_a(&sP[ks * 32], 72);
      v16bf pa1 = load_frag_a(&sP[16 * 72 + ks * 32], 72);
      for (int df = 0; df < 4; ++df) {
        v16bf vb = load_frag_b(vbase + df * 16 * NSEQ + ks * 32, NSEQ);
        of[0][df] = WMMA_BF16(pa0, vb, of[0][df]);
        of[1][df] = WMMA_BF16(pa1, vb, of[1][df]);
      }
    }
  }
  // Normalize and scatter O into (B,N,C) bf16.
  for (int mf = 0; mf < 2; ++mf)
    for (int vr = 0; vr < 8; ++vr) {
      float inv = 1.f / redsum16(rl[mf][vr]);
      int m = qblk * 32 + mf * 16 + vr + 8 * half;
      for (int df = 0; df < 4; ++df) {
        int col = hh * DHEAD + df * 16 + nl;
        obuf[((size_t)b * NSEQ + m) * C_DIM + col] = (__bf16)(of[mf][df][vr] * inv);
      }
    }
}

// ---------------------------------------------------------------------------
extern "C" void kernel_launch(void* const* d_in, const int* in_sizes, int n_in,
                              void* d_out, int out_size, void* d_ws, size_t ws_size,
                              hipStream_t stream) {
  const float* x      = (const float*)d_in[0];
  const float* c      = (const float*)d_in[1];
  const float* qkv_w  = (const float*)d_in[2];
  const float* qkv_b  = (const float*)d_in[3];
  const float* proj_w = (const float*)d_in[4];
  const float* proj_b = (const float*)d_in[5];
  const float* fc1_w  = (const float*)d_in[6];
  const float* fc1_b  = (const float*)d_in[7];
  const float* fc2_w  = (const float*)d_in[8];
  const float* fc2_b  = (const float*)d_in[9];
  const float* ada_w  = (const float*)d_in[10];
  const float* ada_b  = (const float*)d_in[11];
  float* out = (float*)d_out;

  // Workspace carve-up (all 256B-aligned sizes), ~134 MB total.
  char* p = (char*)d_ws;
  float*  ada   = (float*)p;                 p += (size_t)BATCH * ADA_W * 4;          // 147456
  __bf16* wqkv  = (__bf16*)p;                p += (size_t)3 * C_DIM * C_DIM * 2;      // 3538944
  __bf16* wproj = (__bf16*)p;                p += (size_t)C_DIM * C_DIM * 2;          // 1179648
  __bf16* wfc1  = (__bf16*)p;                p += (size_t)HFF * C_DIM * 2;            // 4718592
  __bf16* wfc2  = (__bf16*)p;                p += (size_t)C_DIM * HFF * 2;            // 4718592
  __bf16* hbuf  = (__bf16*)p;                p += (size_t)MROWS * C_DIM * 2;          // 12582912 (also O buffer)
  __bf16* qbuf  = (__bf16*)p;                p += (size_t)MROWS * C_DIM * 2;
  __bf16* kbuf  = (__bf16*)p;                p += (size_t)MROWS * C_DIM * 2;
  __bf16* vtbuf = (__bf16*)p;                p += (size_t)MROWS * C_DIM * 2;
  float*  xbuf1 = (float*)p;                 p += (size_t)MROWS * C_DIM * 4;          // 25165824
  __bf16* h3    = (__bf16*)p;                p += (size_t)MROWS * HFF * 2;            // 50331648

  // 0) weights -> bf16 (once per launch; keeps GEMM loops pure WMMA)
  cvt_bf16_kernel<<<(3 * C_DIM * C_DIM) / 1024, 256, 0, stream>>>(qkv_w,  wqkv,  3 * C_DIM * C_DIM);
  cvt_bf16_kernel<<<(C_DIM * C_DIM) / 1024,     256, 0, stream>>>(proj_w, wproj, C_DIM * C_DIM);
  cvt_bf16_kernel<<<(HFF * C_DIM) / 1024,       256, 0, stream>>>(fc1_w,  wfc1,  HFF * C_DIM);
  cvt_bf16_kernel<<<(C_DIM * HFF) / 1024,       256, 0, stream>>>(fc2_w,  wfc2,  C_DIM * HFF);

  // 1) ada = silu(c) @ ada_w.T + ada_b
  ada_kernel<<<dim3(ADA_W / 256, BATCH), 256, 0, stream>>>(c, ada_w, ada_b, ada);

  // 2) h = modulate(LN(x), sh_msa, sc_msa)  (bf16)
  ln_mod_kernel<<<MROWS, 256, 0, stream>>>(x, ada, 0, C_DIM, hbuf);

  // 3) QKV GEMM (8192 x 2304 x 768), split epilogue
  gemm_kernel<0><<<dim3((3 * C_DIM) / 128, MROWS / 128), 256, 0, stream>>>(
      hbuf, wqkv, qkv_b, 3 * C_DIM, C_DIM,
      nullptr, nullptr, nullptr, nullptr, qbuf, kbuf, vtbuf);

  // 4) Flash attention -> O into hbuf (B,N,C bf16)
  attn_kernel<<<BATCH * NHEAD * (NSEQ / 32), 32, 0, stream>>>(qbuf, kbuf, vtbuf, hbuf);

  // 5) proj GEMM + gated residual: x1 = x + g_msa * (O @ proj_w.T + b)
  gemm_kernel<1><<<dim3(C_DIM / 128, MROWS / 128), 256, 0, stream>>>(
      hbuf, wproj, proj_b, C_DIM, C_DIM,
      xbuf1, nullptr, x, ada + 2 * C_DIM, nullptr, nullptr, nullptr);

  // 6) h = modulate(LN(x1), sh_mlp, sc_mlp)
  ln_mod_kernel<<<MROWS, 256, 0, stream>>>(xbuf1, ada, 3 * C_DIM, 4 * C_DIM, hbuf);

  // 7) fc1 GEMM + exact GELU (8192 x 3072 x 768) -> bf16
  gemm_kernel<2><<<dim3(HFF / 128, MROWS / 128), 256, 0, stream>>>(
      hbuf, wfc1, fc1_b, HFF, C_DIM,
      nullptr, h3, nullptr, nullptr, nullptr, nullptr, nullptr);

  // 8) fc2 GEMM + gated residual (8192 x 768 x 3072) -> d_out (fp32)
  gemm_kernel<1><<<dim3(C_DIM / 128, MROWS / 128), 256, 0, stream>>>(
      h3, wfc2, fc2_b, C_DIM, HFF,
      out, nullptr, xbuf1, ada + 5 * C_DIM, nullptr, nullptr, nullptr);

  (void)in_sizes; (void)n_in; (void)out_size; (void)ws_size; (void)p;
}
